// NetVlad_9861244912107
// MI455X (gfx1250) — matrix operand order repaired and checked by reading, not compiled, not run on hardware
//
#include <hip/hip_runtime.h>
#include <hip/hip_bf16.h>
#include <math.h>

// ---------------------------------------------------------------------------
// NetVLAD fused kernel for MI455X (gfx1250, wave32, WMMA bf16)
//   x:(B,F,N) f32, W:(F,C), b:(C), mu:(C,F)  ->  out:(B, C*F)
//   B=32 F=512 N=4096 C=64
//
// Roofline: x = 256 MB f32 -> ~11 us @ 23.3 TB/s if read exactly once.
// 17.2 GFLOP of GEMM needs matrix pipes (bf16 WMMA, f32 accum) to stay
// memory-bound. Fused kernel: TDM streams x tiles into LDS (overlapped with
// compute), logits+softmax+VLAD accumulate per slice, VLAD partials kept in
// 128 VGPRs/lane and flushed once with f32 global atomics.
// ---------------------------------------------------------------------------

typedef float  v8f   __attribute__((ext_vector_type(8)));
typedef __bf16 v8bf  __attribute__((ext_vector_type(8)));
typedef __bf16 v16bf __attribute__((ext_vector_type(16)));
typedef unsigned int v4u __attribute__((ext_vector_type(4)));
typedef int v4i __attribute__((ext_vector_type(4)));
typedef int v8i __attribute__((ext_vector_type(8)));

#define NVL_B 32
#define NVL_F 512
#define NVL_N 4096
#define NVL_C 64
#define NSLICE 16          // workgroups per batch
#define TN 32              // n-rows per tile (== bf16 WMMA K)

// --- LDS layout (bytes), all 16B aligned ---
#define OFF_STAGE 0        // f32  [512][33]  TDM-staged x tile (padded pitch)
#define OFF_W     67584    // bf16 [64][520]  W transposed [c][f]
#define OFF_XST   134144   // bf16 [32][520]  x tile [n][f]
#define OFF_XS    167424   // bf16 [512][40]  x tile [f][n]
#define OFF_AT    208384   // bf16 [64][40]   a transposed [c][n]
#define OFF_LO    213504   // f32  [32][66]   logits / softmax scratch
#define OFF_ASUM  221952   // f32  [64]
#define LDS_BYTES 222208
#define WP  520
#define XP  520
#define XNP 40
#define ANP 40
#define LOP 66

// ---- fragment loaders (bf16, 16B-aligned LDS reads -> ds_read_b128) -------
__device__ __forceinline__ v16bf ld_afrag(const __bf16* p) {
  // A 16x32 bf16 layout: per lane, elements 0..7 = K+0..7, 8..15 = K+16..23
  v8bf lo8 = *(const v8bf*)p;
  v8bf hi8 = *(const v8bf*)(p + 16);
  return __builtin_shufflevector(lo8, hi8, 0,1,2,3,4,5,6,7,8,9,10,11,12,13,14,15);
}
__device__ __forceinline__ v16bf ld_bfrag(const __bf16* p) {
  // contiguous-K loader (unused; B uses ld_afrag): per lane, 16 contiguous K values
  v8bf lo8 = *(const v8bf*)p;
  v8bf hi8 = *(const v8bf*)(p + 8);
  return __builtin_shufflevector(lo8, hi8, 0,1,2,3,4,5,6,7,8,9,10,11,12,13,14,15);
}

// ---- Tensor Data Mover: 2D tile (tile_dim0=32 n-contig, tile_dim1=512 f),
//      row stride 4096 f32, 4B elems, pad 1 DWORD per 32 DWORDs (pitch 33) ---
__device__ __forceinline__ void tdm_load_x_tile(const float* gsrc, unsigned lds_byte) {
  unsigned long long ga = (unsigned long long)(size_t)gsrc;
  v4u g0 = { 1u,                                   // count=1, user D#
             lds_byte,                              // lds_addr
             (unsigned)(ga & 0xFFFFFFFFull),        // global_addr[31:0]
             (unsigned)((ga >> 32) & 0x01FFFFFFu) | (2u << 30) }; // addr[56:32] | type=2
  v8i g1 = { (int)((2u << 16) | (1u << 20) | (4u << 22)), // data_size=4B, pad_en, pad_int=32dw (amount=1dw)
             (int)(4096u << 16),                    // tensor_dim0=4096 (lo16 in [31:16])
             (int)(512u  << 16),                    // dim0 hi=0 | tensor_dim1=512
             (int)(32u   << 16),                    // dim1 hi=0 | tile_dim0=32
             (int)512,                              // tile_dim1=512, tile_dim2=0
             (int)4096,                             // tensor_dim0_stride lo32
             0, 0 };
  v4i z4 = {0,0,0,0};
  v8i z8 = {0,0,0,0,0,0,0,0};
  // clang-23 / therock-10.0 form: 6 args (extra int32x8 group before cpol)
  __builtin_amdgcn_tensor_load_to_lds(g0, g1, z4, z4, z8, 0);
}

__global__ __launch_bounds__(256)
void netvlad_main(const float* __restrict__ x, const float* __restrict__ W,
                  const float* __restrict__ bias,
                  float* __restrict__ vlad_ws, float* __restrict__ asum_ws)
{
  extern __shared__ char smem[];
  float*  stage = (float*)(smem + OFF_STAGE);
  __bf16* wlds  = (__bf16*)(smem + OFF_W);
  __bf16* xst   = (__bf16*)(smem + OFF_XST);
  __bf16* xs    = (__bf16*)(smem + OFF_XS);
  __bf16* at    = (__bf16*)(smem + OFF_AT);
  float*  lo    = (float*)(smem + OFF_LO);
  float*  asum  = (float*)(smem + OFF_ASUM);

  const int tid  = threadIdx.x;
  const int lane = tid & 31;
  const int wav  = tid >> 5;
  const int h    = lane >> 4;    // lane half (wave32 WMMA layout)
  const int m    = lane & 15;
  const int b    = blockIdx.y;
  const int n_begin = blockIdx.x * (NVL_N / NSLICE);
  const int NITER = (NVL_N / NSLICE) / TN;   // 8

  if (tid < NVL_C) asum[tid] = 0.0f;

  // Stage W: f32 global -> bf16 LDS, transposed [c][f] (padded pitch 520)
  for (int i = tid; i < NVL_F * NVL_C; i += 256) {
    int c = i & 63, f = i >> 6;                  // coalesced over c
    wlds[c * WP + f] = (__bf16)W[f * NVL_C + c];
  }

  const int rt = wav >> 2;                       // logits row tile (n)
  const int ct = wav & 3;                        // logits col tile (c)
  const float bval = bias[ct * 16 + m];

  // VLAD accumulators: 16 tiles * 8 VGPRs = 128 VGPRs/lane, live all kernel
  v8f vacc[16];
  #pragma unroll
  for (int i = 0; i < 16; ++i) {
    v8f z = {0.f,0.f,0.f,0.f,0.f,0.f,0.f,0.f};
    vacc[i] = z;
  }

  const unsigned stage_lds = (unsigned)(size_t)(void*)stage;
  if (wav == 0)
    tdm_load_x_tile(x + (size_t)b * NVL_F * NVL_N + n_begin, stage_lds);

  for (int it = 0; it < NITER; ++it) {
    if (wav == 0) __builtin_amdgcn_s_wait_tensorcnt(0);
    __syncthreads();                              // stage ready; xs/xst/at free

    // f32 stage (pitch 33) -> bf16 in both layouts
    for (int i = tid; i < TN * NVL_F; i += 256) {
      int n = i & 31, f = i >> 5;
      __bf16 v = (__bf16)stage[f * 33 + n];
      xst[n * XP + f]  = v;
      xs[f * XNP + n]  = v;
    }
    __syncthreads();                              // stage consumed

    if (wav == 0 && (it + 1) < NITER)             // overlap next DMA w/ compute
      tdm_load_x_tile(x + (size_t)b * NVL_F * NVL_N + n_begin + (it + 1) * TN,
                      stage_lds);

    // ---- GEMM-1: logits 16x16 tile per wave, K = 512 -------------------
    v8f acc = {0.f,0.f,0.f,0.f,0.f,0.f,0.f,0.f};
    #pragma unroll
    for (int kb = 0; kb < NVL_F; kb += 32) {
      v16bf afrag = ld_afrag(xst  + (rt * 16 + m) * XP + kb + 8  * h);
      v16bf bfrag = ld_afrag(wlds + (ct * 16 + m) * WP + kb + 8  * h);
      acc = __builtin_amdgcn_wmma_f32_16x16x32_bf16(
              false, afrag, false, bfrag, (short)0, acc, false, false);
    }
    #pragma unroll
    for (int v = 0; v < 8; ++v)                   // D layout: M = v + 8h, N = m
      lo[(rt * 16 + v + 8 * h) * LOP + ct * 16 + m] = acc[v] + bval;
    __syncthreads();

    // ---- softmax over C per row (f32 in LDS), emit bf16 a^T ------------
    if (tid < TN) {
      float mx = -3.4e38f;
      for (int c = 0; c < NVL_C; ++c) mx = fmaxf(mx, lo[tid * LOP + c]);
      float s = 0.f;
      for (int c = 0; c < NVL_C; ++c) {
        float e = __expf(lo[tid * LOP + c] - mx);
        lo[tid * LOP + c] = e; s += e;
      }
      float inv = 1.0f / s;
      for (int c = 0; c < NVL_C; ++c) {
        float a = lo[tid * LOP + c] * inv;
        lo[tid * LOP + c] = a;
        at[c * ANP + tid] = (__bf16)a;
      }
    }
    __syncthreads();                              // a (f32) + a^T (bf16) ready

    if (tid < NVL_C) {                            // asum[c] += sum_n a[n][c] (f32)
      float s = 0.f;
      for (int r = 0; r < TN; ++r) s += lo[r * LOP + tid];
      asum[tid] += s;
    }

    // ---- GEMM-2: vlad[c,f] += a^T(c x 32n) * x(32n x f), K = 32 --------
    #pragma unroll
    for (int t = 0; t < 16; ++t) {
      int t2  = wav + t * 8;                      // 128 tiles over 8 waves
      int ct2 = t2 >> 5, ft = t2 & 31;
      v16bf afrag = ld_afrag(at + (ct2 * 16 + m) * ANP + 8  * h);
      v16bf bfrag = ld_afrag(xs + (ft  * 16 + m) * XNP + 8  * h);
      vacc[t] = __builtin_amdgcn_wmma_f32_16x16x32_bf16(
                  false, afrag, false, bfrag, (short)0, vacc[t], false, false);
    }
  }

  // ---- flush partial VLAD + asum with f32 global atomics ---------------
  #pragma unroll
  for (int t = 0; t < 16; ++t) {
    int t2  = wav + t * 8;
    int ct2 = t2 >> 5, ft = t2 & 31;
    int f   = ft * 16 + m;
    #pragma unroll
    for (int v = 0; v < 8; ++v) {
      int c = ct2 * 16 + v + 8 * h;
      unsafeAtomicAdd(&vlad_ws[((size_t)b * NVL_C + c) * NVL_F + f], vacc[t][v]);
    }
  }
  if (tid < NVL_C) unsafeAtomicAdd(&asum_ws[b * NVL_C + tid], asum[tid]);
}

// ---------------------------------------------------------------------------
__global__ __launch_bounds__(256)
void netvlad_zero(float* __restrict__ ws, int n) {
  int i = blockIdx.x * 256 + threadIdx.x;
  if (i < n) ws[i] = 0.0f;
}

// per (b,c): row = vlad - asum*mu; intra L2-normalize; accumulate global sumsq
__global__ __launch_bounds__(256)
void netvlad_rows(const float* __restrict__ vlad_ws, const float* __restrict__ asum_ws,
                  const float* __restrict__ mu, float* __restrict__ out,
                  float* __restrict__ bsq)
{
  const int blk = blockIdx.x;                 // B*C blocks
  const int b = blk >> 6, c = blk & 63;
  const int tid = threadIdx.x, lane = tid & 31, wav = tid >> 5;
  const float s = asum_ws[b * NVL_C + c];
  const size_t base = ((size_t)b * NVL_C + c) * NVL_F;

  float v0 = vlad_ws[base + tid]       - s * mu[c * NVL_F + tid];
  float v1 = vlad_ws[base + tid + 256] - s * mu[c * NVL_F + tid + 256];
  float sq = v0 * v0 + v1 * v1;
  #pragma unroll
  for (int off = 16; off > 0; off >>= 1) sq += __shfl_down(sq, off, 32);

  __shared__ float red[8];
  if (lane == 0) red[wav] = sq;
  __syncthreads();
  if (tid == 0) {
    float t = 0.f;
    for (int i = 0; i < 8; ++i) t += red[i];
    red[0] = t;
  }
  __syncthreads();

  const float nrm = sqrtf(red[0]);
  const float inv = 1.0f / fmaxf(nrm, 1e-12f);
  out[(size_t)b * (NVL_C * NVL_F) + c * NVL_F + tid]       = v0 * inv;
  out[(size_t)b * (NVL_C * NVL_F) + c * NVL_F + tid + 256] = v1 * inv;
  if (tid == 0) {
    float rn = nrm * inv;                     // 1 if ||v||>eps else ||v||/eps
    unsafeAtomicAdd(&bsq[b], rn * rn);
  }
}

__global__ __launch_bounds__(256)
void netvlad_global_norm(float* __restrict__ out, const float* __restrict__ bsq) {
  int i = blockIdx.x * 256 + threadIdx.x;     // B*C*F = 1048576 elements
  int b = i >> 15;
  out[i] *= 1.0f / fmaxf(sqrtf(bsq[b]), 1e-12f);
}

// ---------------------------------------------------------------------------
extern "C" void kernel_launch(void* const* d_in, const int* in_sizes, int n_in,
                              void* d_out, int out_size, void* d_ws, size_t ws_size,
                              hipStream_t stream) {
  const float* x    = (const float*)d_in[0];
  const float* W    = (const float*)d_in[1];
  const float* bias = (const float*)d_in[2];
  const float* mu   = (const float*)d_in[3];
  float* out = (float*)d_out;

  float* ws      = (float*)d_ws;
  float* vlad_ws = ws;                            // 32*64*512   = 1048576
  float* asum_ws = ws + 1048576;                  // 32*64       = 2048
  float* bsq     = ws + 1048576 + 2048;           // 32
  const int nzero = 1048576 + 2048 + 32;

  (void)in_sizes; (void)n_in; (void)out_size; (void)ws_size;

  (void)hipFuncSetAttribute(reinterpret_cast<const void*>(netvlad_main),
                            hipFuncAttributeMaxDynamicSharedMemorySize, LDS_BYTES);

  netvlad_zero<<<(nzero + 255) / 256, 256, 0, stream>>>(ws, nzero);
  netvlad_main<<<dim3(NSLICE, NVL_B), 256, LDS_BYTES, stream>>>(
      x, W, bias, vlad_ws, asum_ws);
  netvlad_rows<<<NVL_B * NVL_C, 256, 0, stream>>>(vlad_ws, asum_ws, mu, out, bsq);
  netvlad_global_norm<<<(NVL_B * NVL_C * NVL_F) / 256, 256, 0, stream>>>(out, bsq);
}
